// EncoderDecoder_29618094473457
// MI455X (gfx1250) — matrix operand order repaired and checked by reading, not compiled, not run on hardware
//
#include <hip/hip_runtime.h>
#include <hip/hip_bf16.h>

#define BB 32
#define SS 128
#define HH 1024
#define G4 4096
#define VV 32000
#define TDEC 32
#define KT_H (HH/32)        // 32 K-tiles of 32
#define NT_V ((VV+15)/16)   // 2000 N-tiles for output proj

typedef __bf16 bf16;
typedef __attribute__((ext_vector_type(16))) __bf16 v16bf;
typedef __attribute__((ext_vector_type(8)))  __bf16 v8bf;
typedef __attribute__((ext_vector_type(8)))  float  v8f;

// ---------------------------------------------------------------------------
// Repack fp32 row-major weights (K x N) into bf16 WMMA-B tiles.
// Tile = 32 K x 16 N, stored as [lane][16 elems] contiguous (32B per lane).
// lane<16 : elem e -> K = k0+(e>>3)*16+(e&7)   , N = n0+lane
// lane>=16: elem e -> K = k0+(e>>3)*16+8+(e&7) , N = n0+lane-16
// Tiles for a fixed nTile are consecutive in kTile (K streams contiguously).
// ---------------------------------------------------------------------------
__global__ void pack_w(const float* __restrict__ W, bf16* __restrict__ out,
                       int K, int N, long total) {
  long idx = (long)blockIdx.x * 256 + threadIdx.x;
  if (idx >= total) return;
  int  e    = (int)(idx & 15);
  int  lane = (int)((idx >> 4) & 31);
  long tile = idx >> 9;
  int  KT   = K >> 5;
  long nTile = tile / KT;
  int  kTile = (int)(tile % KT);
  int  k = kTile * 32 + ((e >> 3) << 4) + ((lane >> 4) << 3) + (e & 7);
  long n = nTile * 16 + (lane & 15);
  float v = (n < N) ? W[(long)k * N + n] : 0.0f;
  out[idx] = (bf16)v;
}

__global__ void init_state(bf16* __restrict__ hbuf, float* __restrict__ c,
                           int* __restrict__ y) {
  int i = blockIdx.x * 256 + threadIdx.x;
  if (i < 2 * BB * HH) hbuf[i] = (bf16)0.0f;
  if (i < BB * HH)     c[i]    = 0.0f;
  if (i < BB)          y[i]    = 1;     // decoder start token
}

// Load 8 fp32 and round to a bf16 A-fragment half.
__device__ inline v8bf cvt8(const float* __restrict__ p) {
  v8f f = *(const v8f*)p;
  v8bf r;
#pragma unroll
  for (int i = 0; i < 8; ++i) r[i] = (bf16)f[i];
  return r;
}

// ---------------------------------------------------------------------------
// One fused LSTM time step with inline embedding gather.
// Grid = 32 WGs (one per 32 hidden cols), 256 thr = 8 waves.
// Wave w: gate = w&3, jtSub = w>>2. Each wave computes BOTH 16-row m-tiles,
// reusing every weight B-tile register load for two WMMAs (halves B traffic).
// z = emb[tok]@Wx + h@Wh (bf16 WMMA, fp32 acc), gates exchanged via LDS,
// fp32 cell update with optional encoder padding mask.
// ---------------------------------------------------------------------------
__launch_bounds__(256)
__global__ void lstm_step(const float* __restrict__ emb,          // vocab x H fp32
                          const int* __restrict__ toks, int tstride, int toff,
                          const bf16* __restrict__ h_in, bf16* __restrict__ h_out,
                          float* __restrict__ c,
                          const bf16* __restrict__ Wx, const bf16* __restrict__ Wh,
                          const float* __restrict__ bias,
                          const int* __restrict__ msrc, int t) {
  __shared__ float zlds[2][4][BB][16];
  const int tid   = threadIdx.x;
  const int lane  = tid & 31;
  const int wave  = tid >> 5;
  const int gate  = wave & 3;
  const int jtSub = wave >> 2;
  const int jt    = blockIdx.x * 2 + jtSub;     // hidden col tile 0..63
  const int nTile = gate * 64 + jt;             // col tile within 4H

  const int mlo  = lane & 15;
  const int koff = (lane >> 4) * 8;             // A layout half-select
  const int tok0 = toks[mlo * tstride + toff];
  const int tok1 = toks[(16 + mlo) * tstride + toff];
  const float* xr0 = emb + (long)tok0 * HH;
  const float* xr1 = emb + (long)tok1 * HH;
  const bf16*  hr0 = h_in + mlo * HH;
  const bf16*  hr1 = h_in + (16 + mlo) * HH;
  const bf16*  wxT = Wx + (long)nTile * KT_H * 512 + lane * 16;
  const bf16*  whT = Wh + (long)nTile * KT_H * 512 + lane * 16;

  v8f acc0 = {}, acc1 = {};
  for (int kt = 0; kt < KT_H; ++kt) {
    v16bf bmat = *(const v16bf*)(wxT + (long)kt * 512);
    if (kt + 1 < KT_H) __builtin_prefetch(wxT + (long)(kt + 1) * 512, 0, 3);
    union { v16bf v; v8bf h[2]; } a0, a1;
    a0.h[0] = cvt8(xr0 + kt * 32 + koff);
    a0.h[1] = cvt8(xr0 + kt * 32 + 16 + koff);
    a1.h[0] = cvt8(xr1 + kt * 32 + koff);
    a1.h[1] = cvt8(xr1 + kt * 32 + 16 + koff);
    acc0 = __builtin_amdgcn_wmma_f32_16x16x32_bf16(false, a0.v, false, bmat,
                                                   (short)0, acc0, false, false);
    acc1 = __builtin_amdgcn_wmma_f32_16x16x32_bf16(false, a1.v, false, bmat,
                                                   (short)0, acc1, false, false);
  }
  for (int kt = 0; kt < KT_H; ++kt) {
    v16bf bmat = *(const v16bf*)(whT + (long)kt * 512);
    if (kt + 1 < KT_H) __builtin_prefetch(whT + (long)(kt + 1) * 512, 0, 3);
    union { v16bf v; v8bf h[2]; } a0, a1;
    a0.h[0] = *(const v8bf*)(hr0 + kt * 32 + koff);
    a0.h[1] = *(const v8bf*)(hr0 + kt * 32 + 16 + koff);
    a1.h[0] = *(const v8bf*)(hr1 + kt * 32 + koff);
    a1.h[1] = *(const v8bf*)(hr1 + kt * 32 + 16 + koff);
    acc0 = __builtin_amdgcn_wmma_f32_16x16x32_bf16(false, a0.v, false, bmat,
                                                   (short)0, acc0, false, false);
    acc1 = __builtin_amdgcn_wmma_f32_16x16x32_bf16(false, a1.v, false, bmat,
                                                   (short)0, acc1, false, false);
  }

  // C/D layout: lanes<16 -> M=r, lanes>=16 -> M=8+r; N = lane&15
  const int n = lane & 15;
#pragma unroll
  for (int r = 0; r < 8; ++r) {
    int mr = r + ((lane >> 4) << 3);
    zlds[jtSub][gate][mr][n]      = acc0[r];
    zlds[jtSub][gate][16 + mr][n] = acc1[r];
  }
  __syncthreads();

  const int j0 = blockIdx.x * 32;
  for (int e = tid; e < BB * 32; e += 256) {
    int   mb = e >> 5, nn = e & 31, j = j0 + nn;
    int   hs = nn >> 4, nl = nn & 15;
    float zi = zlds[hs][0][mb][nl] + bias[j];
    float zf = zlds[hs][1][mb][nl] + bias[HH + j];
    float zg = zlds[hs][2][mb][nl] + bias[2 * HH + j];
    float zo = zlds[hs][3][mb][nl] + bias[3 * HH + j];
    float c_old = c[mb * HH + j];
    float si = 1.0f / (1.0f + expf(-zi));
    float sf = 1.0f / (1.0f + expf(-zf));
    float so = 1.0f / (1.0f + expf(-zo));
    float cn = sf * c_old + si * tanhf(zg);
    float hn = so * tanhf(cn);
    if (msrc && msrc[mb * SS + t] == 0) {       // encoder padding mask
      cn = c_old;
      hn = (float)h_in[mb * HH + j];
    }
    c[mb * HH + j]     = cn;
    h_out[mb * HH + j] = (bf16)hn;
  }
}

// ---------------------------------------------------------------------------
// Output projection: logits[b][v] = h @ W_out + b_out -> d_out[b][t][:].
// Grid = 250 WGs; wave w owns nTile = blockIdx.x*8 + w and BOTH m-tiles,
// reusing each B-tile load for two WMMAs.
// ---------------------------------------------------------------------------
__launch_bounds__(256)
__global__ void proj_out(const bf16* __restrict__ h, const bf16* __restrict__ Wo,
                         const float* __restrict__ bo, float* __restrict__ out, int t) {
  const int lane  = threadIdx.x & 31;
  const int wave  = threadIdx.x >> 5;
  const int nTile = blockIdx.x * 8 + wave;
  if (nTile >= NT_V) return;

  const int mlo  = lane & 15;
  const int koff = (lane >> 4) * 8;
  const bf16* hr0 = h + mlo * HH;
  const bf16* hr1 = h + (16 + mlo) * HH;
  const bf16* wT  = Wo + (long)nTile * KT_H * 512 + lane * 16;

  v8f acc0 = {}, acc1 = {};
  for (int kt = 0; kt < KT_H; ++kt) {
    v16bf bmat = *(const v16bf*)(wT + (long)kt * 512);
    if (kt + 1 < KT_H) __builtin_prefetch(wT + (long)(kt + 1) * 512, 0, 3);
    union { v16bf v; v8bf hh[2]; } a0, a1;
    a0.hh[0] = *(const v8bf*)(hr0 + kt * 32 + koff);
    a0.hh[1] = *(const v8bf*)(hr0 + kt * 32 + 16 + koff);
    a1.hh[0] = *(const v8bf*)(hr1 + kt * 32 + koff);
    a1.hh[1] = *(const v8bf*)(hr1 + kt * 32 + 16 + koff);
    acc0 = __builtin_amdgcn_wmma_f32_16x16x32_bf16(false, a0.v, false, bmat,
                                                   (short)0, acc0, false, false);
    acc1 = __builtin_amdgcn_wmma_f32_16x16x32_bf16(false, a1.v, false, bmat,
                                                   (short)0, acc1, false, false);
  }
  const int col = nTile * 16 + mlo;
  const float bv = bo[col];
#pragma unroll
  for (int r = 0; r < 8; ++r) {
    int b0 = r + ((lane >> 4) << 3);
    out[((long)b0 * TDEC + t) * VV + col]        = acc0[r] + bv;
    out[((long)(16 + b0) * TDEC + t) * VV + col] = acc1[r] + bv;
  }
}

// ---------------------------------------------------------------------------
// Per-row softmax (in place on d_out[b][t][:]) + first-index argmax -> y[b].
// ---------------------------------------------------------------------------
__launch_bounds__(512)
__global__ void softmax_argmax(float* __restrict__ out, int* __restrict__ y, int t) {
  __shared__ float smax[512];
  __shared__ int   sidx[512];
  __shared__ float ssum[512];
  const int b = blockIdx.x, tid = threadIdx.x;
  float* row = out + ((long)b * TDEC + t) * VV;

  float vmax = -3.4e38f; int vidx = 0;
  for (int v = tid; v < VV; v += 512) {
    float xv = row[v];
    if (xv > vmax) { vmax = xv; vidx = v; }
  }
  smax[tid] = vmax; sidx[tid] = vidx;
  __syncthreads();
  for (int s = 256; s > 0; s >>= 1) {
    if (tid < s) {
      float xo = smax[tid + s]; int io = sidx[tid + s];
      if (xo > smax[tid] || (xo == smax[tid] && io < sidx[tid])) {
        smax[tid] = xo; sidx[tid] = io;
      }
    }
    __syncthreads();
  }
  const float mx = smax[0];
  float lsum = 0.0f;
  for (int v = tid; v < VV; v += 512) lsum += expf(row[v] - mx);
  ssum[tid] = lsum;
  __syncthreads();
  for (int s = 256; s > 0; s >>= 1) {
    if (tid < s) ssum[tid] += ssum[tid + s];
    __syncthreads();
  }
  const float inv = 1.0f / ssum[0];
  for (int v = tid; v < VV; v += 512) row[v] = expf(row[v] - mx) * inv;
  if (tid == 0) y[b] = sidx[0];
}

// ---------------------------------------------------------------------------
extern "C" void kernel_launch(void* const* d_in, const int* in_sizes, int n_in,
                              void* d_out, int out_size, void* d_ws, size_t ws_size,
                              hipStream_t stream) {
  (void)in_sizes; (void)n_in; (void)out_size; (void)ws_size;
  const int*   source  = (const int*)  d_in[0];
  const float* emb_enc = (const float*)d_in[1];
  const float* Wx_e    = (const float*)d_in[2];
  const float* Wh_e    = (const float*)d_in[3];
  const float* b_e     = (const float*)d_in[4];
  const float* emb_dec = (const float*)d_in[5];
  const float* Wx_d    = (const float*)d_in[6];
  const float* Wh_d    = (const float*)d_in[7];
  const float* b_d     = (const float*)d_in[8];
  const float* W_out   = (const float*)d_in[9];
  const float* b_out   = (const float*)d_in[10];
  float* out = (float*)d_out;

  char* ws = (char*)d_ws;
  size_t off = 0;
  auto carve = [&](size_t bytes) -> char* {
    char* p = ws + off;
    off = (off + bytes + 255) & ~(size_t)255;
    return p;
  };
  bf16* pWx_e = (bf16*)carve((size_t)HH * G4 * 2);
  bf16* pWh_e = (bf16*)carve((size_t)HH * G4 * 2);
  bf16* pWx_d = (bf16*)carve((size_t)HH * G4 * 2);
  bf16* pWh_d = (bf16*)carve((size_t)HH * G4 * 2);
  bf16* pWo   = (bf16*)carve((size_t)HH * VV * 2);
  bf16* hbuf  = (bf16*)carve((size_t)2 * BB * HH * 2);   // double-buffered h
  float* cst  = (float*)carve((size_t)BB * HH * 4);
  int*  ytok  = (int*) carve((size_t)BB * 4);

  // One-shot fp32 -> bf16 tiled repack (~96 MB, L2-resident afterwards)
  const long nG = (long)HH * G4;
  pack_w<<<(int)((nG + 255) / 256), 256, 0, stream>>>(Wx_e, pWx_e, HH, G4, nG);
  pack_w<<<(int)((nG + 255) / 256), 256, 0, stream>>>(Wh_e, pWh_e, HH, G4, nG);
  pack_w<<<(int)((nG + 255) / 256), 256, 0, stream>>>(Wx_d, pWx_d, HH, G4, nG);
  pack_w<<<(int)((nG + 255) / 256), 256, 0, stream>>>(Wh_d, pWh_d, HH, G4, nG);
  const long nO = (long)HH * VV;
  pack_w<<<(int)((nO + 255) / 256), 256, 0, stream>>>(W_out, pWo, HH, VV, nO);
  init_state<<<(2 * BB * HH + 255) / 256, 256, 0, stream>>>(hbuf, cst, ytok);

  // Encoder: 128 masked LSTM steps (embedding gather fused into the step)
  for (int t = 0; t < SS; ++t) {
    bf16* h_in  = hbuf + (size_t)(t & 1) * BB * HH;
    bf16* h_out = hbuf + (size_t)((t + 1) & 1) * BB * HH;
    lstm_step<<<32, 256, 0, stream>>>(emb_enc, source, SS, t, h_in, h_out, cst,
                                      pWx_e, pWh_e, b_e, source, t);
  }
  // Decoder: 32 greedy steps
  for (int td = 0; td < TDEC; ++td) {
    int t = SS + td;
    bf16* h_in  = hbuf + (size_t)(t & 1) * BB * HH;
    bf16* h_out = hbuf + (size_t)((t + 1) & 1) * BB * HH;
    lstm_step<<<32, 256, 0, stream>>>(emb_dec, ytok, 1, 0, h_in, h_out, cst,
                                      pWx_d, pWh_d, b_d, nullptr, 0);
    proj_out<<<NT_V / 8, 256, 0, stream>>>(h_out, pWo, b_out, out, td);
    softmax_argmax<<<BB, 512, 0, stream>>>(out, ytok, td);
  }
}
